// Attention_29832842838194
// MI455X (gfx1250) — matrix operand order repaired and checked by reading, not compiled, not run on hardware
//
#include <hip/hip_runtime.h>

typedef __attribute__((ext_vector_type(16))) _Float16 v16h;
typedef __attribute__((ext_vector_type(8)))  float    v8f;

#define B_     2
#define N_     2048
#define DIM_   1024
#define HEADS_ 8
#define DHEAD_ 64
#define INNER_ 512
#define MTOT   (B_ * N_)      // 4096
#define QKVC   (3 * INNER_)   // 1536

// ---------------------------------------------------------------------------
// WMMA fragment loaders (wave32; A/B use contiguous k order, C/D per ISA 7.12.2)
// A 16x32 f16: lane L holds row M=L&15; elem j -> K = j + 16*(L>>4)
// B 32x16 f16 from LDS stored TRANSPOSED [N][K]: lane holds col N=L&15;
//   elem j -> K = j + 16*(L>>4)  (16 contiguous halves per lane)
// C/D f32: VGPR r -> row M = r + 8*(L>>4), col N = L&15
// ---------------------------------------------------------------------------
__device__ __forceinline__ v16h load_a_frag(const _Float16* p, int ld) {
  const int lane = threadIdx.x & 31;
  const _Float16* q = p + (lane & 15) * ld + (lane >> 4) * 16;
  v16h a;
#pragma unroll
  for (int j = 0; j < 8; ++j) a[j] = q[j];
#pragma unroll
  for (int j = 0; j < 8; ++j) a[8 + j] = q[8 + j];
  return a;
}

__device__ __forceinline__ v16h load_b_frag(const _Float16* p, int ld) {
  const int lane = threadIdx.x & 31;
  const _Float16* q = p + (lane & 15) * ld + (lane >> 4) * 16;
  v16h b;
#pragma unroll
  for (int j = 0; j < 16; ++j) b[j] = q[j];
  return b;
}

// Async DMA one 16B segment: global (64-bit VA) -> LDS (addrspace(3) offset =
// low 32 bits of the generic pointer). Tracked by ASYNCcnt.
__device__ __forceinline__ void async_copy16(const _Float16* gsrc, _Float16* lds_dst) {
  const uint64_t ga = (uint64_t)(uintptr_t)gsrc;
  const uint32_t la = (uint32_t)(uintptr_t)lds_dst;
  asm volatile("global_load_async_to_lds_b128 %0, %1, off" :: "v"(la), "v"(ga) : "memory");
}
__device__ __forceinline__ void async_wait0() {
  asm volatile("s_wait_asynccnt 0x0" ::: "memory");
}

// ---------------------------------------------------------------------------
// 1. LayerNorm over the SEQUENCE axis (dim=1): stats per (b, d) column.
// ---------------------------------------------------------------------------
__global__ void ln_kernel(const float* __restrict__ x, const float* __restrict__ g,
                          _Float16* __restrict__ xn) {
  const int b = blockIdx.x >> 2;
  const int d = ((blockIdx.x & 3) << 8) + threadIdx.x;
  const float* xp = x + (size_t)b * N_ * DIM_ + d;
  float s = 0.f, ss = 0.f;
  for (int n = 0; n < N_; ++n) {
    float v = xp[(size_t)n * DIM_];
    s += v; ss += v * v;
  }
  const float mean = s * (1.0f / N_);
  const float var  = ss * (1.0f / N_) - mean * mean;
  const float inv  = rsqrtf(var + 1e-5f) * g[d];
  _Float16* op = xn + (size_t)b * N_ * DIM_ + d;
  for (int n = 0; n < N_; ++n) {
    float v = xp[(size_t)n * DIM_];
    op[(size_t)n * DIM_] = (_Float16)((v - mean) * inv);
  }
}

// ---------------------------------------------------------------------------
// 2. Cast weights to f16; fuse wq|wkv into one [1024 x 1536] matrix.
// ---------------------------------------------------------------------------
__global__ void cast_w_kernel(const float* __restrict__ wq, const float* __restrict__ wkv,
                              const float* __restrict__ wo,
                              _Float16* __restrict__ wqkv, _Float16* __restrict__ woh) {
  const int i = blockIdx.x * 256 + threadIdx.x;
  if (i < DIM_ * QKVC) {
    const int row = i / QKVC, c = i % QKVC;
    const float v = (c < INNER_) ? wq[(size_t)row * INNER_ + c]
                                 : wkv[(size_t)row * (2 * INNER_) + (c - INNER_)];
    wqkv[i] = (_Float16)v;
  }
  if (i < INNER_ * DIM_) woh[i] = (_Float16)wo[i];
}

// ---------------------------------------------------------------------------
// 3./5. Tiled WMMA GEMM:  C[M x Nc] = A[M x K] @ B[K x Nc], f16 in, f32 acc.
//    128 threads = 4 waves; block tile 64x64; K-step 64 (8 WMMA / barrier pair).
//    A tile staged by async DMA (ASYNCcnt); B staged manually (transpose).
//    MODE 0: scatter into q (x0.125) / k / v head-major buffers (f16).
//    MODE 1: write f32 to out.
// ---------------------------------------------------------------------------
template <int MODE>
__global__ void __launch_bounds__(128, 1)
gemm_kernel(const _Float16* __restrict__ A, const _Float16* __restrict__ Bm,
            int K, int Nc,
            _Float16* __restrict__ qw, _Float16* __restrict__ kw,
            _Float16* __restrict__ vw, float* __restrict__ outf) {
  __shared__ __align__(16) _Float16 sA[64 * 72];   // [m][k], ld=72 (144B rows, 16B aligned)
  __shared__ __align__(16) _Float16 sB[64 * 72];   // transposed [n][k], ld=72

  const int t    = threadIdx.x;
  const int lane = t & 31;
  const int wave = t >> 5;
  const int m0 = blockIdx.y * 64;
  const int n0 = blockIdx.x * 64;
  const int wm = (wave >> 1) * 32;
  const int wn = (wave & 1) * 32;

  v8f c[2][2] = {};

  for (int k0 = 0; k0 < K; k0 += 64) {
    // stage A 64x64 via async global->LDS DMA (8 KB, 4 x b128 per thread)
#pragma unroll
    for (int i = 0; i < 4; ++i) {
      const int u = t + i * 128;            // 512 x 16B segments
      const int row = u >> 3, seg = u & 7;
      async_copy16(A + (size_t)(m0 + row) * K + k0 + seg * 8, &sA[row * 72 + seg * 8]);
    }
    // stage B 64x64 transposed -> sB[n][k] (manual: DMA cannot transpose)
    const _Float16* gb = Bm + (size_t)k0 * Nc + n0;
#pragma unroll
    for (int i = 0; i < 16; ++i) {
      const int u = t + i * 128;            // 2048 half-pairs
      const int kr = u >> 5, nc2 = u & 31;
      const _Float16* gp = gb + (size_t)kr * Nc + nc2 * 2;
      sB[(nc2 * 2) * 72 + kr]     = gp[0];
      sB[(nc2 * 2 + 1) * 72 + kr] = gp[1];
    }
    // prefetch next K tile into L2 while we compute this one
    if (k0 + 64 < K) {
      __builtin_prefetch(A + (size_t)(m0 + (t >> 1)) * K + k0 + 64 + (t & 1) * 32, 0, 3);
      __builtin_prefetch(Bm + (size_t)(k0 + 64 + (t >> 1)) * Nc + n0 + (t & 1) * 32, 0, 3);
    }
    async_wait0();
    __syncthreads();

#pragma unroll
    for (int kc = 0; kc < 2; ++kc) {
      v16h a0 = load_a_frag(&sA[wm * 72 + kc * 32], 72);
      v16h a1 = load_a_frag(&sA[(wm + 16) * 72 + kc * 32], 72);
      v16h b0 = load_b_frag(&sB[wn * 72 + kc * 32], 72);
      v16h b1 = load_b_frag(&sB[(wn + 16) * 72 + kc * 32], 72);
      c[0][0] = __builtin_amdgcn_wmma_f32_16x16x32_f16(false, a0, false, b0, (short)0, c[0][0], false, false);
      c[0][1] = __builtin_amdgcn_wmma_f32_16x16x32_f16(false, a0, false, b1, (short)0, c[0][1], false, false);
      c[1][0] = __builtin_amdgcn_wmma_f32_16x16x32_f16(false, a1, false, b0, (short)0, c[1][0], false, false);
      c[1][1] = __builtin_amdgcn_wmma_f32_16x16x32_f16(false, a1, false, b1, (short)0, c[1][1], false, false);
    }
    __syncthreads();
  }

  const int hi = lane >> 4, ln = lane & 15;
#pragma unroll
  for (int i = 0; i < 2; ++i)
#pragma unroll
    for (int j = 0; j < 2; ++j)
#pragma unroll
      for (int r = 0; r < 8; ++r) {
        const int row = m0 + wm + i * 16 + r + 8 * hi;
        const int col = n0 + wn + j * 16 + ln;
        const float val = c[i][j][r];
        if (MODE == 0) {
          const int b = row >> 11, n = row & (N_ - 1);
          if (col < INNER_) {
            const int h = col >> 6, d = col & 63;
            qw[(((size_t)(b * HEADS_ + h) * N_) + n) * DHEAD_ + d] = (_Float16)(val * 0.125f);
          } else if (col < 2 * INNER_) {
            const int c2 = col - INNER_, h = c2 >> 6, d = c2 & 63;
            kw[(((size_t)(b * HEADS_ + h) * N_) + n) * DHEAD_ + d] = (_Float16)val;
          } else {
            const int c2 = col - 2 * INNER_, h = c2 >> 6, d = c2 & 63;
            vw[(((size_t)(b * HEADS_ + h) * N_) + n) * DHEAD_ + d] = (_Float16)val;
          }
        } else {
          outf[(size_t)row * DIM_ + col] = val;
        }
      }
}

// ---------------------------------------------------------------------------
// 4. Flash attention. grid = (N/64, B*H). 128 threads = 4 waves.
//    Block: 64 query rows; wave: 16 query rows. Key blocks of 64, online softmax.
//    Q/K tiles staged by async DMA; V staged manually (transposed in LDS).
// ---------------------------------------------------------------------------
__global__ void __launch_bounds__(128, 1)
attn_kernel(const _Float16* __restrict__ qw, const _Float16* __restrict__ kw,
            const _Float16* __restrict__ vw, _Float16* __restrict__ ao) {
  __shared__ __align__(16) _Float16 sQ[64 * 72];    // [query][d]
  __shared__ __align__(16) _Float16 sK[64 * 72];    // [key][d] == transposed-B [N=key][K=d]
  __shared__ __align__(16) _Float16 sV[64 * 72];    // [d][key] == transposed-B [N=d][K=key]
  __shared__ __align__(16) _Float16 sP[4][16 * 72]; // per-wave P staging [row][key]

  const int t    = threadIdx.x;
  const int lane = t & 31;
  const int wave = t >> 5;
  const int bh = blockIdx.y;
  const int q0 = blockIdx.x * 64;
  const _Float16* qbase = qw + (size_t)bh * N_ * DHEAD_;
  const _Float16* kbase = kw + (size_t)bh * N_ * DHEAD_;
  const _Float16* vbase = vw + (size_t)bh * N_ * DHEAD_;

  // stage Q tile (64x64) via async DMA
#pragma unroll
  for (int i = 0; i < 4; ++i) {
    const int u = t + i * 128;
    const int row = u >> 3, seg = u & 7;
    async_copy16(qbase + (size_t)(q0 + row) * DHEAD_ + seg * 8, &sQ[row * 72 + seg * 8]);
  }
  async_wait0();
  __syncthreads();

  // Q fragments are loop-invariant: load once (q pre-scaled by 1/sqrt(dhead))
  const v16h qa0 = load_a_frag(&sQ[(wave * 16) * 72], 72);
  const v16h qa1 = load_a_frag(&sQ[(wave * 16) * 72 + 32], 72);

  const int hi = lane >> 4, ln = lane & 15;
  v8f o[4] = {};
  float mrow[8], lrow[8];
#pragma unroll
  for (int r = 0; r < 8; ++r) { mrow[r] = -1e30f; lrow[r] = 0.f; }

  for (int kb = 0; kb < N_; kb += 64) {
    __syncthreads();                       // prior iteration's reads done
    // K tile via async DMA (layout-preserving)
#pragma unroll
    for (int i = 0; i < 4; ++i) {
      const int u = t + i * 128;
      const int row = u >> 3, seg = u & 7;
      async_copy16(kbase + (size_t)(kb + row) * DHEAD_ + seg * 8, &sK[row * 72 + seg * 8]);
    }
    // V tile transposed -> sV[d][key] (manual)
#pragma unroll
    for (int i = 0; i < 16; ++i) {
      const int u = t + i * 128;
      const int row = u >> 5, col = u & 31;
      const _Float16* gp = vbase + (size_t)(kb + row) * DHEAD_ + col * 2;
      sV[(col * 2) * 72 + row]     = gp[0];
      sV[(col * 2 + 1) * 72 + row] = gp[1];
    }
    // prefetch next key block into L2
    if (kb + 64 < N_) {
      __builtin_prefetch(kbase + (size_t)(kb + 64 + (t >> 1)) * DHEAD_ + (t & 1) * 32, 0, 3);
      __builtin_prefetch(vbase + (size_t)(kb + 64 + (t >> 1)) * DHEAD_ + (t & 1) * 32, 0, 3);
    }
    async_wait0();
    __syncthreads();

    // S(16x64) = Qwave @ K^T
    v8f s[4] = {};
#pragma unroll
    for (int tt = 0; tt < 4; ++tt) {
      v16h b0 = load_b_frag(&sK[(tt * 16) * 72], 72);
      v16h b1 = load_b_frag(&sK[(tt * 16) * 72 + 32], 72);
      s[tt] = __builtin_amdgcn_wmma_f32_16x16x32_f16(false, qa0, false, b0, (short)0, s[tt], false, false);
      s[tt] = __builtin_amdgcn_wmma_f32_16x16x32_f16(false, qa1, false, b1, (short)0, s[tt], false, false);
    }

    // online softmax per row (row = r + 8*hi lives in the 16 lanes sharing hi)
#pragma unroll
    for (int r = 0; r < 8; ++r) {
      float mx = fmaxf(fmaxf(s[0][r], s[1][r]), fmaxf(s[2][r], s[3][r]));
#pragma unroll
      for (int off = 8; off >= 1; off >>= 1) mx = fmaxf(mx, __shfl_xor(mx, off, 32));
      const float mnew  = fmaxf(mrow[r], mx);
      const float alpha = __expf(mrow[r] - mnew);
      mrow[r] = mnew;
      float rs = 0.f;
#pragma unroll
      for (int tt = 0; tt < 4; ++tt) {
        const float p = __expf(s[tt][r] - mnew);
        s[tt][r] = p;
        rs += p;
      }
#pragma unroll
      for (int off = 8; off >= 1; off >>= 1) rs += __shfl_xor(rs, off, 32);
      lrow[r] = lrow[r] * alpha + rs;
#pragma unroll
      for (int j = 0; j < 4; ++j) o[j][r] *= alpha;
    }

    // C-layout P -> A-layout via per-wave LDS staging
    _Float16* pb = &sP[wave][0];
#pragma unroll
    for (int tt = 0; tt < 4; ++tt)
#pragma unroll
      for (int r = 0; r < 8; ++r)
        pb[(r + 8 * hi) * 72 + tt * 16 + ln] = (_Float16)s[tt][r];
    asm volatile("s_wait_dscnt 0x0" ::: "memory");

    v16h pa0 = load_a_frag(pb, 72);
    v16h pa1 = load_a_frag(pb + 32, 72);
#pragma unroll
    for (int j = 0; j < 4; ++j) {
      v16h b0 = load_b_frag(&sV[(j * 16) * 72], 72);
      v16h b1 = load_b_frag(&sV[(j * 16) * 72 + 32], 72);
      o[j] = __builtin_amdgcn_wmma_f32_16x16x32_f16(false, pa0, false, b0, (short)0, o[j], false, false);
      o[j] = __builtin_amdgcn_wmma_f32_16x16x32_f16(false, pa1, false, b1, (short)0, o[j], false, false);
    }
  }

  // normalize and write merged-head output [B*N][INNER]
  const int b = bh >> 3, h = bh & 7;
#pragma unroll
  for (int r = 0; r < 8; ++r) {
    const float rinv = 1.0f / lrow[r];
    const int n = q0 + wave * 16 + r + 8 * hi;
#pragma unroll
    for (int j = 0; j < 4; ++j) {
      const int d = h * 64 + j * 16 + ln;
      ao[((size_t)b * N_ + n) * INNER_ + d] = (_Float16)(o[j][r] * rinv);
    }
  }
}

// ---------------------------------------------------------------------------
extern "C" void kernel_launch(void* const* d_in, const int* in_sizes, int n_in,
                              void* d_out, int out_size, void* d_ws, size_t ws_size,
                              hipStream_t stream) {
  const float* x   = (const float*)d_in[0];
  const float* g   = (const float*)d_in[1];
  const float* wq  = (const float*)d_in[2];
  const float* wkv = (const float*)d_in[3];
  const float* wo  = (const float*)d_in[4];
  float* out = (float*)d_out;

  _Float16* ws = (_Float16*)d_ws;
  size_t off = 0;
  _Float16* xn   = ws + off; off += (size_t)MTOT * DIM_;
  _Float16* wqkv = ws + off; off += (size_t)DIM_ * QKVC;
  _Float16* woh  = ws + off; off += (size_t)INNER_ * DIM_;
  _Float16* qw   = ws + off; off += (size_t)B_ * HEADS_ * N_ * DHEAD_;
  _Float16* kw   = ws + off; off += (size_t)B_ * HEADS_ * N_ * DHEAD_;
  _Float16* vw   = ws + off; off += (size_t)B_ * HEADS_ * N_ * DHEAD_;
  _Float16* ao   = ws + off; off += (size_t)MTOT * INNER_;

  ln_kernel<<<B_ * (DIM_ / 256), 256, 0, stream>>>(x, g, xn);
  cast_w_kernel<<<(DIM_ * QKVC) / 256, 256, 0, stream>>>(wq, wkv, wo, wqkv, woh);
  gemm_kernel<0><<<dim3(QKVC / 64, MTOT / 64), 128, 0, stream>>>(
      xn, wqkv, DIM_, QKVC, qw, kw, vw, nullptr);
  attn_kernel<<<dim3(N_ / 64, B_ * HEADS_), 128, 0, stream>>>(qw, kw, vw, ao);
  gemm_kernel<1><<<dim3(DIM_ / 64, MTOT / 64), 128, 0, stream>>>(
      ao, woh, INNER_, DIM_, nullptr, nullptr, nullptr, out);
}
